// DynamicConv_12541304504536
// MI455X (gfx1250) — matrix operand order repaired and checked by reading, not compiled, not run on hardware
//
#include <hip/hip_runtime.h>
#include <hip/hip_bf16.h>
#include <math.h>

typedef __attribute__((ext_vector_type(16))) _Float16 v16h;
typedef __attribute__((ext_vector_type(8)))  float    v8f;

#define NBATCH 8
#define NCH    64
#define HH     128
#define WW     128
#define PLANE  (HH * WW)         // 16384
#define NJ     576               // C*K*K
#define EPSBN  1e-5f

// ---------------------------------------------------------------------------
// Kernel 1: global average pool.  One block per (n,c) plane, float4 loads.
// ---------------------------------------------------------------------------
__global__ __launch_bounds__(256) void gap_kernel(const float* __restrict__ x,
                                                  float* __restrict__ f) {
    __shared__ float red[256];
    const int plane = blockIdx.x;
    const float4* xp = (const float4*)(x + (size_t)plane * PLANE);
    float s = 0.0f;
    for (int i = threadIdx.x; i < PLANE / 4; i += 256) {
        float4 v = xp[i];
        s += v.x + v.y + v.z + v.w;
    }
    red[threadIdx.x] = s;
    __syncthreads();
    for (int off = 128; off > 0; off >>= 1) {
        if (threadIdx.x < off) red[threadIdx.x] += red[threadIdx.x + off];
        __syncthreads();
    }
    if (threadIdx.x == 0) f[plane] = red[0] * (1.0f / (float)PLANE);
}

// ---------------------------------------------------------------------------
// Kernel 2: dynamic-filter generation.  One wave32.
//   g = f(8x64) @ W^T(64x576) via v_wmma_f32_16x16x32_f16 (K split 0..31/32..63)
//   then BN (eval) and softmax over the 9 kernel positions.
// A layout (16-bit 16x32): lane half -> kbase 0/16; elem e -> K=kbase+e.
//                          M = lane%16.
// B layout (16-bit 32x16): lanes 0-15 K=0..15 (elem e -> K=e),
//                          lanes 16-31 K=16..31.      N = lane%16.
// D layout: VGPR r, lanes 0-15 -> M=r (our 8 valid batch rows), N = lane.
// ---------------------------------------------------------------------------
__global__ __launch_bounds__(32) void filter_kernel(
    const float* __restrict__ f,       // (8,64)
    const float* __restrict__ conv_w,  // (576,64)
    const float* __restrict__ gamma,
    const float* __restrict__ beta,
    const float* __restrict__ mean,
    const float* __restrict__ var,
    float* __restrict__ wout)          // (8,64,9)
{
    __shared__ float g[NBATCH * NJ];   // 18 KB
    const int lane  = threadIdx.x;     // 0..31 (one wave)
    const int m     = lane & 15;
    const int kbA   = (lane < 16) ? 0 : 16;
    const int kbB   = (lane < 16) ? 0 : 16;

    // Pack A (batch rows >= 8 are zero so the padded rows contribute nothing)
    v16h a0, a1;
    #pragma unroll
    for (int e = 0; e < 16; ++e) {
        int k = kbA + e;
        float v0 = (m < NBATCH) ? f[m * 64 + k]      : 0.0f;
        float v1 = (m < NBATCH) ? f[m * 64 + 32 + k] : 0.0f;
        a0[e] = (_Float16)v0;
        a1[e] = (_Float16)v1;
    }

    for (int jt = 0; jt < NJ / 16; ++jt) {     // 36 N-tiles
        const int j = jt * 16 + m;             // column of W^T == row of conv_w
        v16h b0, b1;
        #pragma unroll
        for (int e = 0; e < 16; ++e) {
            int k = kbB + e;
            b0[e] = (_Float16)conv_w[j * 64 + k];        // B[k][n] = W[j][k]
            b1[e] = (_Float16)conv_w[j * 64 + 32 + k];
        }
        v8f acc = {};
        acc = __builtin_amdgcn_wmma_f32_16x16x32_f16(false, a0, false, b0,
                                                     (short)0, acc, false, false);
        acc = __builtin_amdgcn_wmma_f32_16x16x32_f16(false, a1, false, b1,
                                                     (short)0, acc, false, false);
        if (lane < 16) {                       // lanes 0-15 hold M=0..7 in acc[0..7]
            const int jj    = jt * 16 + lane;
            const float scl = gamma[jj] * rsqrtf(var[jj] + EPSBN);
            const float mu  = mean[jj];
            const float bt  = beta[jj];
            #pragma unroll
            for (int r = 0; r < NBATCH; ++r)
                g[r * NJ + jj] = (acc[r] - mu) * scl + bt;
        }
    }
    __syncthreads();

    // softmax over the 9 taps for each of the 8*64 (n,c) groups
    for (int grp = lane; grp < NBATCH * NCH; grp += 32) {
        const int   nn = grp >> 6, cc = grp & 63;
        const float* gp = &g[nn * NJ + cc * 9];
        float mx = gp[0];
        #pragma unroll
        for (int k = 1; k < 9; ++k) mx = fmaxf(mx, gp[k]);
        float e9[9], s = 0.0f;
        #pragma unroll
        for (int k = 0; k < 9; ++k) { e9[k] = __expf(gp[k] - mx); s += e9[k]; }
        const float inv = 1.0f / s;
        float* op = &wout[grp * 9];
        #pragma unroll
        for (int k = 0; k < 9; ++k) op[k] = e9[k] * inv;
    }
}

// ---------------------------------------------------------------------------
// Kernel 3: reflect-pad 3x3 dynamic weighted sum + residual.
// Block = 128 threads = one 4-row strip of one (n,c) plane.
// Stage 6x130 input rows in LDS; each thread emits a float4 of out and x-out.
// ---------------------------------------------------------------------------
__global__ __launch_bounds__(128) void apply_kernel(
    const float* __restrict__ x,
    const float* __restrict__ w9,      // (n*C, 9)
    float* __restrict__ out,
    float* __restrict__ res)
{
    __shared__ float lds[6][132];
    const int tile  = blockIdx.x & 31;         // 32 strips of 4 rows
    const int plane = blockIdx.x >> 5;         // n*C planes
    const int y0    = tile << 2;
    const float* xp = x + (size_t)plane * PLANE;

    // load rows y0-1 .. y0+4, cols -1 .. 128 with reflect(1)
    for (int i = threadIdx.x; i < 6 * 130; i += 128) {
        int r  = i / 130, cc = i % 130;
        int gy = y0 - 1 + r;
        gy = (gy < 0) ? -gy : ((gy > HH - 1) ? 2 * (HH - 1) - gy : gy);
        int gx = cc - 1;
        gx = (gx < 0) ? -gx : ((gx > WW - 1) ? 2 * (WW - 1) - gx : gx);
        lds[r][cc] = xp[gy * WW + gx];
    }
    float wgt[9];
    const float* wp = w9 + plane * 9;
    #pragma unroll
    for (int k = 0; k < 9; ++k) wgt[k] = wp[k];
    __syncthreads();

    const int ry = threadIdx.x >> 5;           // row within strip (0..3)
    const int p0 = (threadIdx.x & 31) << 2;    // starting pixel (x4 vector)
    float4 o, r4;
    #pragma unroll
    for (int q = 0; q < 4; ++q) {
        const int p = p0 + q;
        float acc = 0.0f;
        #pragma unroll
        for (int ki = 0; ki < 3; ++ki)
            #pragma unroll
            for (int kj = 0; kj < 3; ++kj)
                acc = fmaf(wgt[ki * 3 + kj], lds[ry + ki][p + kj], acc);
        const float xv = lds[ry + 1][p + 1];
        (&o.x)[q]  = acc;
        (&r4.x)[q] = xv - acc;
    }
    const size_t base = (size_t)plane * PLANE + (size_t)(y0 + ry) * WW + p0;
    *(float4*)(out + base) = o;
    *(float4*)(res + base) = r4;
}

// ---------------------------------------------------------------------------
extern "C" void kernel_launch(void* const* d_in, const int* in_sizes, int n_in,
                              void* d_out, int out_size, void* d_ws, size_t ws_size,
                              hipStream_t stream) {
    const float* x      = (const float*)d_in[0];
    const float* conv_w = (const float*)d_in[1];
    const float* gamma  = (const float*)d_in[2];
    const float* beta   = (const float*)d_in[3];
    const float* mean   = (const float*)d_in[4];
    const float* var    = (const float*)d_in[5];

    float* ws = (float*)d_ws;
    float* f  = ws;                       // 512 floats
    float* w9 = ws + NBATCH * NCH;        // 4608 floats

    float* out = (float*)d_out;
    float* res = out + (size_t)NBATCH * NCH * PLANE;

    gap_kernel<<<NBATCH * NCH, 256, 0, stream>>>(x, f);
    filter_kernel<<<1, 32, 0, stream>>>(f, conv_w, gamma, beta, mean, var, w9);
    apply_kernel<<<NBATCH * NCH * (HH / 4), 128, 0, stream>>>(x, w9, out, res);
}
